// GATModelWithKL_6339371729010
// MI455X (gfx1250) — compile-verified, matrix operand order checked by reading, not run on hardware
//
#include <hip/hip_runtime.h>

typedef __attribute__((ext_vector_type(16))) _Float16 v16h;
typedef __attribute__((ext_vector_type(8)))  _Float16 v8h;
typedef __attribute__((ext_vector_type(8)))  float    v8f;

// Per-lane K index for 16-bit WMMA A/B fragments (CDNA5 ISA 7.12.2):
// lanes 0-15: K in {0..7} u {16..23}; lanes 16-31: K in {8..15} u {24..31}.
__device__ __forceinline__ int wmma_k_of(int lane, int idx) {
  int base = (lane & 16) ? 8 : 0;
  return base + ((idx < 8) ? idx : idx + 8);
}

// ---------------------------------------------------------------------------
// K0: swizzle f32 weights [K x N] into per-lane WMMA B-fragment order (f16).
// out index = (((ctile*(K/32)+kstep)*32 + lane)*16 + idx)
// ---------------------------------------------------------------------------
__global__ void swizzle_w_kernel(const float* __restrict__ W, _Float16* __restrict__ out,
                                 int K, int N) {
  int t = blockIdx.x * blockDim.x + threadIdx.x;
  int ks = K >> 5;                       // K/32
  int total = (N >> 4) * ks * 32 * 16;
  if (t >= total) return;
  int idx   = t & 15;
  int lane  = (t >> 4) & 31;
  int q     = t >> 9;
  int kstep = q % ks;
  int ctile = q / ks;
  int k = kstep * 32 + wmma_k_of(lane, idx);
  int n = ctile * 16 + (lane & 15);
  out[t] = (_Float16)W[(size_t)k * N + n];
}

// ---------------------------------------------------------------------------
// K1: Gaussian heads. grid = (256 row-chunks, 8 modalities), 256 threads.
// Weight-stationary: W_m (mean+lv, 128KB) staged in LDS once per block.
// Writes mp16/lv16 [m][row][e] (row = b*8+s) as f16.
// ---------------------------------------------------------------------------
__global__ void gauss_kernel(const float* __restrict__ priv,
                             const float* __restrict__ mW, const float* __restrict__ mB,
                             const float* __restrict__ lW, const float* __restrict__ lB,
                             _Float16* __restrict__ mp16, _Float16* __restrict__ lv16) {
  extern __shared__ char smem[];
  float* Wm = (float*)smem;          // 128*128
  float* Wl = Wm + 16384;            // 128*128
  float* Pv = Wl + 16384;            // 256 rows * 128
  int m = blockIdx.y;
  int rowBase = blockIdx.x * 256;
  int t = threadIdx.x;
  for (int i = t; i < 16384; i += 256) {
    Wm[i] = mW[(size_t)m * 16384 + i];
    Wl[i] = lW[(size_t)m * 16384 + i];
  }
  const float* ps = priv + ((size_t)m * 65536 + rowBase) * 128;
  for (int i = t; i < 32768; i += 256) Pv[i] = ps[i];
  __syncthreads();

  int e = t & 127, rh = t >> 7;
  float bm = mB[m * 128 + e], bl = lB[m * 128 + e];
  for (int kb = 0; kb < 32; ++kb) {
    // rows rh+8kb, +2, +4, +6 (this thread owns parity rh)
    const float* p0 = Pv + (rh + 8 * kb) * 128;
    float a0 = 0, a1 = 0, a2 = 0, a3 = 0, b0 = 0, b1 = 0, b2 = 0, b3 = 0;
    for (int d = 0; d < 128; ++d) {
      float wm = Wm[d * 128 + e], wl = Wl[d * 128 + e];
      float q0 = p0[d], q1 = p0[d + 256], q2 = p0[d + 512], q3 = p0[d + 768];
      a0 += q0 * wm; a1 += q1 * wm; a2 += q2 * wm; a3 += q3 * wm;
      b0 += q0 * wl; b1 += q1 * wl; b2 += q2 * wl; b3 += q3 * wl;
    }
    size_t o = ((size_t)m * 65536 + rowBase + rh + 8 * kb) * 128 + e;
    mp16[o]       = (_Float16)(a0 + bm);  lv16[o]       = (_Float16)(b0 + bl);
    mp16[o + 256] = (_Float16)(a1 + bm);  lv16[o + 256] = (_Float16)(b1 + bl);
    mp16[o + 512] = (_Float16)(a2 + bm);  lv16[o + 512] = (_Float16)(b2 + bl);
    mp16[o + 768] = (_Float16)(a3 + bm);  lv16[o + 768] = (_Float16)(b3 + bl);
  }
}

// ---------------------------------------------------------------------------
// K2: pairwise KL + adjacency. One block per b (8192 blocks), 256 threads.
// kl(i->j) = 0.5*(sumlv_j - sumlv_i + sum_d[vp_i*ivq_j + (mp_i-mp_j)^2*ivq_j] - D)
// ---------------------------------------------------------------------------
__global__ void kl_adj_kernel(const _Float16* __restrict__ mp16,
                              const _Float16* __restrict__ lv16,
                              int* __restrict__ adj) {
  extern __shared__ char smem[];
  _Float16* mpS = (_Float16*)smem;        // 8192 halves
  float* vpS = (float*)(mpS + 8192);      // 8192
  float* iqS = vpS + 8192;                // 8192
  float* slv = iqS + 8192;                // 64
  float* klS = slv + 64;                  // 512
  float* klm = klS + 512;                 // 64
  int b = blockIdx.x, t = threadIdx.x;

  if (t < 64) slv[t] = 0.f;
  __syncthreads();
  for (int i = t; i < 8192; i += 256) {        // i = (m*8+s)*128 + e
    int m = i >> 10;
    int rest = i & 1023;
    size_t g = ((size_t)m * 65536 + (size_t)b * 8) * 128 + rest;
    mpS[i] = mp16[g];
    float lv = (float)lv16[g];
    vpS[i] = __expf(lv);
    iqS[i] = __expf(-lv);
    atomicAdd(&slv[i >> 7], lv);
  }
  __syncthreads();
  for (int q = 0; q < 2; ++q) {
    int idx = t + q * 256;                     // (i,j,s), s fastest
    int s = idx & 7, j = (idx >> 3) & 7, i = idx >> 6;
    int oi = (i * 8 + s) * 128, oj = (j * 8 + s) * 128;
    float acc = 0.f;
    for (int d = 0; d < 128; ++d) {
      float mi = (float)mpS[oi + d], mj = (float)mpS[oj + d];
      float dm = mi - mj;
      acc += (vpS[oi + d] + dm * dm) * iqS[oj + d];
    }
    klS[(i * 8 + j) * 8 + s] = 0.5f * (slv[j * 8 + s] - slv[i * 8 + s] + acc - 128.f);
  }
  __syncthreads();
  if (t < 64) {
    float acc = 0.f;
    for (int s = 0; s < 8; ++s) acc += klS[t * 8 + s];
    klm[t] = acc * 0.125f;
  }
  __syncthreads();
  if (t < 64) {
    int i = t >> 3, j = t & 7;
    bool e1 = (i < j) && (klm[i * 8 + j] > 0.5f);
    bool e2 = (j < i) && (klm[j * 8 + i] > 0.5f);
    adj[(size_t)b * 64 + t] = (i == j || e1 || e2) ? 1 : 0;
  }
}

// ---------------------------------------------------------------------------
// K3: GAT layer 1 fused (GEMM 128x512 tile via WMMA f16 + attention in LDS).
// One block = 16 graphs (128 rows). 8 waves; wave w owns row-tile w.
// ---------------------------------------------------------------------------
__global__ void gat1_kernel(const float* __restrict__ x, const _Float16* __restrict__ w1s,
                            const float* __restrict__ asrcW, const float* __restrict__ adstW,
                            const float* __restrict__ bias, const int* __restrict__ adj,
                            _Float16* __restrict__ x1) {
  extern __shared__ char smem[];
  _Float16* xA = (_Float16*)smem;          // 128 x (256+8)
  _Float16* hS = xA + 128 * 264;           // 128 x (512+8)
  float* aS = (float*)(hS + 128 * 520);    // 512
  float* aD = aS + 512;                    // 512
  float* alp = aD + 512;                   // 16*8*8*4
  int t = threadIdx.x;
  int lane = t & 31, wid = t >> 5;
  size_t rowBase = (size_t)blockIdx.x * 128;

  // load x tile, f32 -> f16
  for (int i = t; i < 128 * 256; i += 256) {
    int r = i >> 8, c = i & 255;
    xA[r * 264 + c] = (_Float16)x[(rowBase + r) * 256 + c];
  }
  __syncthreads();

  // GEMM: h = x @ W1  (K=256: 8 k-steps; N=512: 32 col-tiles)
  {
    int arow = wid * 16 + (lane & 15);
    int kbase = (lane & 16) ? 8 : 0;
    int rbase2 = wid * 16 + ((lane & 16) ? 8 : 0);
    for (int ct = 0; ct < 32; ++ct) {
      v8f acc = {};
      for (int ks = 0; ks < 8; ++ks) {
        const _Float16* ap = xA + arow * 264 + ks * 32 + kbase;
        v8h lo = *(const v8h*)(ap);
        v8h hi = *(const v8h*)(ap + 16);
        v16h a = __builtin_shufflevector(lo, hi, 0,1,2,3,4,5,6,7,8,9,10,11,12,13,14,15);
        v16h bf = *(const v16h*)(w1s + ((size_t)(ct * 8 + ks) * 32 + lane) * 16);
        acc = __builtin_amdgcn_wmma_f32_16x16x32_f16(false, a, false, bf,
                                                     (short)0, acc, false, false);
      }
      int col = ct * 16 + (lane & 15);
      for (int r = 0; r < 8; ++r)
        hS[(rbase2 + r) * 520 + col] = (_Float16)acc[r];
    }
  }
  __syncthreads();

  // attention source/target terms: 512 (bl,node,h)
  for (int q = 0; q < 2; ++q) {
    int idx = t + q * 256;
    int h = idx & 3, node = (idx >> 2) & 7, bl = idx >> 5;
    const _Float16* hp = hS + (bl * 8 + node) * 520 + h * 128;
    float s1 = 0.f, s2 = 0.f;
    for (int c = 0; c < 128; ++c) {
      float hv = (float)hp[c];
      s1 += hv * asrcW[h * 128 + c];
      s2 += hv * adstW[h * 128 + c];
    }
    aS[idx] = s1; aD[idx] = s2;
  }
  __syncthreads();

  // alpha softmax over masked sources j: 512 (bl,i,h)
  for (int q = 0; q < 2; ++q) {
    int idx = t + q * 256;
    int h = idx & 3, i = (idx >> 2) & 7, bl = idx >> 5;
    const int* ad = adj + ((size_t)blockIdx.x * 16 + bl) * 64 + i * 8;
    float di = aD[bl * 32 + i * 4 + h];
    float ev[8], mx = -3.0e38f;
    for (int j = 0; j < 8; ++j) {
      float e = di + aS[bl * 32 + j * 4 + h];
      e = (e > 0.f) ? e : 0.2f * e;
      e = ad[j] ? e : -1.0e9f;
      ev[j] = e; mx = fmaxf(mx, e);
    }
    float sum = 0.f;
    for (int j = 0; j < 8; ++j) { ev[j] = __expf(ev[j] - mx); sum += ev[j]; }
    float inv = 1.f / sum;
    for (int j = 0; j < 8; ++j) alp[((bl * 8 + i) * 8 + j) * 4 + h] = ev[j] * inv;
  }
  __syncthreads();

  // out = alpha @ h, +bias, relu, write f16
  for (int oi = 0; oi < 256; ++oi) {
    int o = oi * 256 + t;               // (bl,i,h,c), c fastest
    int c = o & 127, h = (o >> 7) & 3, i = (o >> 9) & 7, bl = o >> 12;
    const float* ap2 = alp + ((bl * 8 + i) * 8) * 4 + h;
    const _Float16* hb = hS + (bl * 8) * 520 + h * 128 + c;
    float v = 0.f;
    for (int j = 0; j < 8; ++j) v += ap2[j * 4] * (float)hb[j * 520];
    v += bias[h * 128 + c];
    v = fmaxf(v, 0.f);
    x1[(rowBase + bl * 8 + i) * 512 + h * 128 + c] = (_Float16)v;
  }
}

// ---------------------------------------------------------------------------
// K4: GAT layer 2 fused (K=512, N=256, H=1, C=256). Output f32 to d_out.
// ---------------------------------------------------------------------------
__global__ void gat2_kernel(const _Float16* __restrict__ x1, const _Float16* __restrict__ w2s,
                            const float* __restrict__ asrcW, const float* __restrict__ adstW,
                            const float* __restrict__ bias, const int* __restrict__ adj,
                            float* __restrict__ out) {
  extern __shared__ char smem[];
  _Float16* xA = (_Float16*)smem;          // 128 x (512+8)
  _Float16* hS = xA + 128 * 520;           // 128 x (256+8)
  float* aS = (float*)(hS + 128 * 264);    // 128
  float* aD = aS + 128;                    // 128
  float* alp = aD + 128;                   // 16*8*8
  int t = threadIdx.x;
  int lane = t & 31, wid = t >> 5;
  size_t rowBase = (size_t)blockIdx.x * 128;

  for (int i = t; i < 128 * 512; i += 256) {
    int r = i >> 9, c = i & 511;
    xA[r * 520 + c] = x1[(rowBase + r) * 512 + c];
  }
  __syncthreads();

  // GEMM: K=512 -> 16 k-steps; N=256 -> 16 col-tiles
  {
    int arow = wid * 16 + (lane & 15);
    int kbase = (lane & 16) ? 8 : 0;
    int rbase2 = wid * 16 + ((lane & 16) ? 8 : 0);
    for (int ct = 0; ct < 16; ++ct) {
      v8f acc = {};
      for (int ks = 0; ks < 16; ++ks) {
        const _Float16* ap = xA + arow * 520 + ks * 32 + kbase;
        v8h lo = *(const v8h*)(ap);
        v8h hi = *(const v8h*)(ap + 16);
        v16h a = __builtin_shufflevector(lo, hi, 0,1,2,3,4,5,6,7,8,9,10,11,12,13,14,15);
        v16h bf = *(const v16h*)(w2s + ((size_t)(ct * 16 + ks) * 32 + lane) * 16);
        acc = __builtin_amdgcn_wmma_f32_16x16x32_f16(false, a, false, bf,
                                                     (short)0, acc, false, false);
      }
      int col = ct * 16 + (lane & 15);
      for (int r = 0; r < 8; ++r)
        hS[(rbase2 + r) * 264 + col] = (_Float16)acc[r];
    }
  }
  __syncthreads();

  if (t < 128) {                         // (bl,node): H=1, C=256
    const _Float16* hp = hS + t * 264;
    float s1 = 0.f, s2 = 0.f;
    for (int c = 0; c < 256; ++c) {
      float hv = (float)hp[c];
      s1 += hv * asrcW[c];
      s2 += hv * adstW[c];
    }
    aS[t] = s1; aD[t] = s2;
  }
  __syncthreads();

  if (t < 128) {                         // alpha: (bl,i)
    int i = t & 7, bl = t >> 3;
    const int* ad = adj + ((size_t)blockIdx.x * 16 + bl) * 64 + i * 8;
    float di = aD[t];
    float ev[8], mx = -3.0e38f;
    for (int j = 0; j < 8; ++j) {
      float e = di + aS[bl * 8 + j];
      e = (e > 0.f) ? e : 0.2f * e;
      e = ad[j] ? e : -1.0e9f;
      ev[j] = e; mx = fmaxf(mx, e);
    }
    float sum = 0.f;
    for (int j = 0; j < 8; ++j) { ev[j] = __expf(ev[j] - mx); sum += ev[j]; }
    float inv = 1.f / sum;
    for (int j = 0; j < 8; ++j) alp[t * 8 + j] = ev[j] * inv;
  }
  __syncthreads();

  for (int oi = 0; oi < 128; ++oi) {
    int o = oi * 256 + t;                // (bl,i,c), c fastest
    int c = o & 255, i = (o >> 8) & 7, bl = o >> 11;
    const float* ap2 = alp + (bl * 8 + i) * 8;
    const _Float16* hb = hS + (bl * 8) * 264 + c;
    float v = 0.f;
    for (int j = 0; j < 8; ++j) v += ap2[j] * (float)hb[j * 264];
    v += bias[c];
    out[(rowBase + bl * 8 + i) * 256 + c] = v;
  }
}

// ---------------------------------------------------------------------------
extern "C" void kernel_launch(void* const* d_in, const int* in_sizes, int n_in,
                              void* d_out, int out_size, void* d_ws, size_t ws_size,
                              hipStream_t stream) {
  (void)in_sizes; (void)n_in; (void)out_size; (void)ws_size;
  const float* x    = (const float*)d_in[0];
  const float* priv = (const float*)d_in[1];
  const float* mW   = (const float*)d_in[2];
  const float* mB   = (const float*)d_in[3];
  const float* lW   = (const float*)d_in[4];
  const float* lB   = (const float*)d_in[5];
  const float* g1W  = (const float*)d_in[6];
  const float* g1as = (const float*)d_in[7];
  const float* g1ad = (const float*)d_in[8];
  const float* g1b  = (const float*)d_in[9];
  const float* g2W  = (const float*)d_in[10];
  const float* g2as = (const float*)d_in[11];
  const float* g2ad = (const float*)d_in[12];
  const float* g2b  = (const float*)d_in[13];
  float* out = (float*)d_out;

  char* ws = (char*)d_ws;
  size_t off = 0;
  auto take = [&](size_t bytes) -> char* {
    char* p = ws + off;
    off = (off + bytes + 255) & ~(size_t)255;
    return p;
  };
  _Float16* mp16 = (_Float16*)take((size_t)8 * 65536 * 128 * 2);  // 128 MB
  _Float16* lv16 = (_Float16*)take((size_t)8 * 65536 * 128 * 2);  // 128 MB
  int*      adj  = (int*)take((size_t)8192 * 64 * 4);             // 2 MB
  _Float16* w1s  = (_Float16*)take((size_t)256 * 512 * 2);
  _Float16* w2s  = (_Float16*)take((size_t)512 * 256 * 2);
  _Float16* x1   = mp16;  // alias: mp16 region is dead after kl_adj_kernel

  swizzle_w_kernel<<<512, 256, 0, stream>>>(g1W, w1s, 256, 512);
  swizzle_w_kernel<<<512, 256, 0, stream>>>(g2W, w2s, 512, 256);
  gauss_kernel<<<dim3(256, 8), 256, 262144, stream>>>(priv, mW, mB, lW, lB, mp16, lv16);
  kl_adj_kernel<<<8192, 256, 84480, stream>>>(mp16, lv16, adj);
  gat1_kernel<<<512, 256, 221184, stream>>>(x, w1s, g1as, g1ad, g1b, adj, x1);
  gat2_kernel<<<512, 256, 205824, stream>>>(x1, w2s, g2as, g2ad, g2b, adj, out);
}
